// ConvLSTMForecast_57947698758168
// MI455X (gfx1250) — compile-verified, matrix operand order checked by reading, not run on hardware
//
#include <hip/hip_runtime.h>
#include <hip/hip_bf16.h>

typedef __attribute__((ext_vector_type(16))) _Float16 v16h;
typedef __attribute__((ext_vector_type(8)))  _Float16 v8h;
typedef __attribute__((ext_vector_type(8)))  float    v8f;
typedef __attribute__((ext_vector_type(4)))  int      v4i;

#define HH   32
#define WWID 32
#define MB   2
#define TX   6
#define INC  11
#define POS  (MB*HH*WWID)   // 2048 spatial positions across batch

#if __has_builtin(__builtin_amdgcn_global_load_async_to_lds_b128)
#define USE_ASYNC_STAGE 1
#else
#define USE_ASYNC_STAGE 0
#endif

typedef __attribute__((address_space(1))) v4i* gptr_v4i;
typedef __attribute__((address_space(3))) v4i* lptr_v4i;

__device__ __forceinline__ void wait_async_zero() {
#if __has_builtin(__builtin_amdgcn_s_wait_asynccnt)
    __builtin_amdgcn_s_wait_asynccnt(0);
#else
    asm volatile("s_wait_asynccnt 0x0" ::: "memory");
#endif
}

// ---------------------------------------------------------------------------
// Weight pre-swizzle: OIHW f32 -> per-lane WMMA A-fragment order, f16.
// Layout: [((mi*25 + tap)*KS + ks)*32 + lane]*16 + e   (16 halves / lane)
// For 16-bit A 16x32 fragments: lane l holds row (l&15);
//   K = ((l>>4)*8) + (e<8 ? e : e+8)   (per CDNA5 ISA 7.12.2)
// ---------------------------------------------------------------------------
template<int COUT, int CINP, int CINR>
__global__ __launch_bounds__(256) void swizzle_w(const float* __restrict__ w,
                                                 _Float16* __restrict__ out)
{
    constexpr int KS = CINP / 32;
    const int total = (COUT/16) * 25 * KS * 512;
    int i = blockIdx.x * blockDim.x + threadIdx.x;
    if (i >= total) return;
    int e    = i & 15;
    int lane = (i >> 4) & 31;
    int rest = i >> 9;
    int ks  = rest % KS; rest /= KS;
    int tap = rest % 25;
    int mi  = rest / 25;
    int row = mi*16 + (lane & 15);
    int K   = ((lane >> 4) << 3) + (e < 8 ? e : e + 8);
    int cin = ks*32 + K;
    float v = 0.f;
    if (cin < CINR) {
        int dy = tap / 5, dx = tap % 5;
        v = w[(((size_t)row*CINR + cin)*5 + dy)*5 + dx];
    }
    out[i] = (_Float16)v;
}

// ---------------------------------------------------------------------------
// Implicit-GEMM 5x5 conv via WMMA with LDS-staged activations.
// act: NHWC f16 [MB][H][W][CIN] (CIN padded to mult of 32, zero pad channels).
// Each block (8 waves) owns one (batch, y) output row; it stages the 5 input
// rows y-2..y+2 with a +-2 x halo (zero-filled) into LDS once — using async
// global->LDS copies (ASYNCcnt) when the toolchain exposes them — then all
// waves stream B fragments from LDS (bank-conflict-padded) while A fragments
// come pre-swizzled from L2-resident global memory. pre: [pos][COUT] f32.
// ---------------------------------------------------------------------------
template<int COUT, int CIN>
__global__ __launch_bounds__(256) void conv_gemm(
    const _Float16* __restrict__ act, const _Float16* __restrict__ wA,
    const float* __restrict__ bias, float* __restrict__ pre)
{
    constexpr int KS   = CIN / 32;
    constexpr int NM2  = COUT / 32;
    constexpr int BPR  = NM2 / 8;      // blocks per (batch,y) row
    constexpr int XSTR = CIN + 8;      // padded LDS stride per x position
    __shared__ _Float16 lds[5 * 36 * XSTR];

    const int tid  = threadIdx.x;
    const int lane = tid & 31;
    const int wave = tid >> 5;
    const int nn   = blockIdx.x / BPR;         // 0..63 : batch*32 + y
    const int mm   = (blockIdx.x % BPR) * 8 + wave;
    const int bm   = nn >> 5;
    const int y    = nn & 31;
    const int jx   = lane & 15;
    const int sub  = (lane >> 4) << 3;

    // ---- cooperative stage: 5 rows x 36 x-positions x CIN channels --------
    {
        const int cvecs = CIN / 8;
        const int chunks = 5 * 36 * cvecs;
        const v8h z = {};
        for (int idx = tid; idx < chunks; idx += 256) {
            int cc   = idx % cvecs;
            int rest = idx / cvecs;
            int xx   = rest % 36;
            int ry   = rest / 36;
            int py = y + ry - 2, px = xx - 2;
            bool ok = (py >= 0) && (py < HH) && (px >= 0) && (px < WWID);
            const _Float16* src = act + (size_t)((bm*HH + py)*WWID + px)*CIN + cc*8;
            _Float16* dst = &lds[(ry*36 + xx)*XSTR + cc*8];
#if USE_ASYNC_STAGE
            if (ok) {
                __builtin_amdgcn_global_load_async_to_lds_b128(
                    (gptr_v4i)src, (lptr_v4i)dst, 0, 0);
            } else {
                *(v8h*)dst = z;   // zero-fill halo (disjoint from async targets)
            }
#else
            v8h v = z;
            if (ok) v = *(const v8h*)src;
            *(v8h*)dst = v;
#endif
        }
#if USE_ASYNC_STAGE
        wait_async_zero();
#endif
    }
    __syncthreads();

    v8f acc00 = {}, acc01 = {}, acc10 = {}, acc11 = {};

    const _Float16* wbase0 = wA + (size_t)(2*mm + 0) * 25 * KS * 512 + lane*16;
    const _Float16* wbase1 = wA + (size_t)(2*mm + 1) * 25 * KS * 512 + lane*16;

    for (int dy = 0; dy < 5; ++dy) {
        const _Float16* lrow = &lds[(dy*36)*XSTR];
#pragma unroll
        for (int dx = 0; dx < 5; ++dx) {
            const int tap = dy*5 + dx;
            const _Float16* l0  = lrow + (jx + dx)*XSTR + sub;
            const _Float16* l1  = l0 + 16*XSTR;
            const _Float16* wa0 = wbase0 + (size_t)tap * KS * 512;
            const _Float16* wa1 = wbase1 + (size_t)tap * KS * 512;
            if (tap < 24) {   // pull next tap's A fragments toward the WGP
                __builtin_prefetch(wa0 + KS*512, 0, 3);
                __builtin_prefetch(wa1 + KS*512, 0, 3);
            }
#pragma unroll
            for (int ks = 0; ks < KS; ++ks) {
                union { v16h v; v8h h[2]; } b0, b1;
                b0.h[0] = *(const v8h*)(l0 + ks*32);
                b0.h[1] = *(const v8h*)(l0 + ks*32 + 16);
                b1.h[0] = *(const v8h*)(l1 + ks*32);
                b1.h[1] = *(const v8h*)(l1 + ks*32 + 16);
                v16h a0 = *(const v16h*)(wa0 + ks*512);
                v16h a1 = *(const v16h*)(wa1 + ks*512);
                acc00 = __builtin_amdgcn_wmma_f32_16x16x32_f16(false, a0, false, b0.v, (short)0, acc00, false, false);
                acc01 = __builtin_amdgcn_wmma_f32_16x16x32_f16(false, a0, false, b1.v, (short)0, acc01, false, false);
                acc10 = __builtin_amdgcn_wmma_f32_16x16x32_f16(false, a1, false, b0.v, (short)0, acc10, false, false);
                acc11 = __builtin_amdgcn_wmma_f32_16x16x32_f16(false, a1, false, b1.v, (short)0, acc11, false, false);
            }
        }
    }

    // Epilogue: C/D layout -> VGPR r: lanes 0-15 row r, lanes 16-31 row r+8
    const int pos0 = bm*1024 + y*WWID + jx;
    const int pos1 = pos0 + 16;
    const int mb0  = mm*32, mb1 = mm*32 + 16;
    const int rsel = (lane < 16) ? 0 : 8;
#pragma unroll
    for (int r = 0; r < 8; ++r) {
        int row = r + rsel;
        float bi0 = bias[mb0 + row];
        float bi1 = bias[mb1 + row];
        pre[(size_t)pos0*COUT + mb0 + row] = acc00[r] + bi0;
        pre[(size_t)pos1*COUT + mb0 + row] = acc01[r] + bi0;
        pre[(size_t)pos0*COUT + mb1 + row] = acc10[r] + bi1;
        pre[(size_t)pos1*COUT + mb1 + row] = acc11[r] + bi1;
    }
}

// ---------------------------------------------------------------------------
// LSTM gates: pre [pos][4*HC] order (i,f,o,g); c [pos][HC] f32 (updated);
// writes h (f16) into channel slices of up to two NHWC buffers.
// ---------------------------------------------------------------------------
template<int HC>
__global__ __launch_bounds__(256) void gate_kernel(
    const float* __restrict__ pre, float* __restrict__ c,
    _Float16* __restrict__ dstA, int offA, int strideA,
    _Float16* __restrict__ dstB, int offB, int strideB)
{
    int i = blockIdx.x * blockDim.x + threadIdx.x;
    if (i >= POS*HC) return;
    int pos = i / HC, ch = i % HC;
    const float* p = pre + (size_t)pos * 4 * HC;
    float ki = p[ch], kf = p[HC + ch], ko = p[2*HC + ch], kg = p[3*HC + ch];
    float si = 1.f / (1.f + __expf(-ki));
    float sf = 1.f / (1.f + __expf(-kf));
    float so = 1.f / (1.f + __expf(-ko));
    float cn = sf * c[i] + si * tanhf(kg);
    float hn = so * tanhf(cn);
    c[i] = cn;
    dstA[(size_t)pos*strideA + offA + ch] = (_Float16)hn;
    if (dstB) dstB[(size_t)pos*strideB + offB + ch] = (_Float16)hn;
}

// x_t (f32 NCHW) -> channels [0,11) of layer-1 NHWC f16 buffer (stride 96)
__global__ __launch_bounds__(256) void ingest_kernel(const float* __restrict__ X,
                                                     int t, _Float16* __restrict__ buf1)
{
    int i = blockIdx.x * blockDim.x + threadIdx.x;
    if (i >= POS*INC) return;
    int pos = i / INC, ch = i % INC;
    int b = pos >> 10, yx = pos & 1023;
    float v = X[(((size_t)(b*TX + t)*INC + ch) << 10) + yx];
    buf1[(size_t)pos*96 + ch] = (_Float16)v;
}

// meo head: 1x1 conv 128->5 on h3 (channels [128,256) of buf3)
__global__ __launch_bounds__(256) void meo_kernel(
    const _Float16* __restrict__ buf3, const float* __restrict__ Wm,
    const float* __restrict__ bmv, float* __restrict__ out, int t)
{
    int i = blockIdx.x * blockDim.x + threadIdx.x;
    if (i >= POS*5) return;
    int pos = i / 5, oc = i % 5;
    const _Float16* h = buf3 + (size_t)pos*256 + 128;
    const float* w = Wm + oc*128;
    float s = bmv[oc];
#pragma unroll 8
    for (int k = 0; k < 128; ++k) s += (float)h[k] * w[k];
    int b = pos >> 10, yx = pos & 1023;
    out[((size_t)b*TX + t)*5120 + oc*1024 + yx] = s;
}

// aqi head stage 1: per (station, batch) 3x3x128 patch -> 256 hidden (tanh)
__global__ __launch_bounds__(256) void aqi_hidden_kernel(
    const _Float16* __restrict__ buf3, const int* __restrict__ locs,
    const float* __restrict__ Wf1, const float* __restrict__ bf1,
    float* __restrict__ hid)
{
    int i = blockIdx.x * blockDim.x + threadIdx.x;
    if (i >= 35*MB*256) return;
    int j = i % 256; int sb = i / 256; int b = sb % MB; int s = sb / MB;
    int ly = locs[2*s], lx = locs[2*s + 1];
    float acc = bf1[j];
    for (int c = 0; c < 128; ++c)
        for (int py = 0; py < 3; ++py)
#pragma unroll
            for (int px = 0; px < 3; ++px) {
                int pos = b*1024 + (ly + py)*WWID + (lx + px);
                float f = (float)buf3[(size_t)pos*256 + 128 + c];
                acc += f * Wf1[(size_t)(c*9 + py*3 + px)*256 + j];
            }
    hid[i] = tanhf(acc);
}

// aqi head stage 2: 256 -> 6, scatter to (m, Tx, S*6) layout
__global__ __launch_bounds__(256) void aqi_out_kernel(
    const float* __restrict__ hid, const float* __restrict__ Wf2,
    const float* __restrict__ bf2, float* __restrict__ out, int t)
{
    int i = blockIdx.x * blockDim.x + threadIdx.x;
    if (i >= 35*MB*6) return;
    int j = i % 6; int sb = i / 6; int b = sb % MB; int s = sb / MB;
    const float* h = hid + (size_t)(s*MB + b)*256;
    float acc = bf2[j];
#pragma unroll 8
    for (int k = 0; k < 256; ++k) acc += h[k] * Wf2[k*6 + j];
    out[((size_t)b*TX + t)*210 + s*6 + j] = acc;
}

// ---------------------------------------------------------------------------
extern "C" void kernel_launch(void* const* d_in, const int* in_sizes, int n_in,
                              void* d_out, int out_size, void* d_ws, size_t ws_size,
                              hipStream_t stream)
{
    (void)in_sizes; (void)n_in; (void)out_size; (void)ws_size;
    const float* X   = (const float*)d_in[0];
    const float* W1  = (const float*)d_in[1];
    const float* b1  = (const float*)d_in[2];
    const float* W2  = (const float*)d_in[3];
    const float* b2  = (const float*)d_in[4];
    const float* W3  = (const float*)d_in[5];
    const float* b3  = (const float*)d_in[6];
    const float* Wm  = (const float*)d_in[7];
    const float* bm  = (const float*)d_in[8];
    const float* Wf1 = (const float*)d_in[9];
    const float* bf1 = (const float*)d_in[10];
    const float* Wf2 = (const float*)d_in[11];
    const float* bf2 = (const float*)d_in[12];
    const int*  locs = (const int*)d_in[13];

    float* aqi_out = (float*)d_out;          // 2*6*210 = 2520 floats
    float* meo_out = aqi_out + 2520;         // 2*6*5120 = 61440 floats

    char* ws = (char*)d_ws;
    size_t off = 0;
    auto alloc = [&](size_t bytes) {
        char* p = ws + off; off += (bytes + 255) & ~(size_t)255; return p;
    };
    _Float16* buf1 = (_Float16*)alloc((size_t)POS*96*2);   // [x(11)|h1(64)|pad(21)]
    _Float16* buf2 = (_Float16*)alloc((size_t)POS*192*2);  // [h1(64)|h2(128)]
    _Float16* buf3 = (_Float16*)alloc((size_t)POS*256*2);  // [h2(128)|h3(128)]
    float* c1 = (float*)alloc((size_t)POS*64*4);
    float* c2 = (float*)alloc((size_t)POS*128*4);
    float* c3 = (float*)alloc((size_t)POS*128*4);
    size_t zero_bytes = off;                               // bufs + cell states
    float* pre1 = (float*)alloc((size_t)POS*256*4);
    float* pre2 = (float*)alloc((size_t)POS*512*4);
    float* pre3 = (float*)alloc((size_t)POS*512*4);
    _Float16* wA1 = (_Float16*)alloc((size_t)16*25*3*512*2);
    _Float16* wA2 = (_Float16*)alloc((size_t)32*25*6*512*2);
    _Float16* wA3 = (_Float16*)alloc((size_t)32*25*8*512*2);
    float* hid = (float*)alloc((size_t)35*MB*256*4);

    // reset h/c state (padding channels of buf1 stay zero all launch)
    (void)hipMemsetAsync(d_ws, 0, zero_bytes, stream);

    // pre-swizzle weights into WMMA A-fragment order (f16)
    { int tot = 16*25*3*512; swizzle_w<256, 96, 75><<<(tot+255)/256,256,0,stream>>>(W1, wA1); }
    { int tot = 32*25*6*512; swizzle_w<512,192,192><<<(tot+255)/256,256,0,stream>>>(W2, wA2); }
    { int tot = 32*25*8*512; swizzle_w<512,256,256><<<(tot+255)/256,256,0,stream>>>(W3, wA3); }

    for (int t = 0; t < TX; ++t) {
        ingest_kernel<<<(POS*INC + 255)/256, 256, 0, stream>>>(X, t, buf1);

        conv_gemm<256, 96><<< 64*(256/32/8), 256, 0, stream>>>(buf1, wA1, b1, pre1);
        gate_kernel<64><<<(POS*64 + 255)/256, 256, 0, stream>>>(
            pre1, c1, buf1, 11, 96, buf2, 0, 192);

        conv_gemm<512,192><<<64*(512/32/8), 256, 0, stream>>>(buf2, wA2, b2, pre2);
        gate_kernel<128><<<(POS*128 + 255)/256, 256, 0, stream>>>(
            pre2, c2, buf2, 64, 192, buf3, 0, 256);

        conv_gemm<512,256><<<64*(512/32/8), 256, 0, stream>>>(buf3, wA3, b3, pre3);
        gate_kernel<128><<<(POS*128 + 255)/256, 256, 0, stream>>>(
            pre3, c3, buf3, 128, 256, (_Float16*)nullptr, 0, 0);

        meo_kernel<<<(POS*5 + 255)/256, 256, 0, stream>>>(buf3, Wm, bm, meo_out, t);
        aqi_hidden_kernel<<<(35*MB*256 + 255)/256, 256, 0, stream>>>(buf3, locs, Wf1, bf1, hid);
        aqi_out_kernel<<<(35*MB*6 + 255)/256, 256, 0, stream>>>(hid, Wf2, bf2, aqi_out, t);
    }
}
